// Rwkv6SelfAttention_35390530519756
// MI455X (gfx1250) — matrix operand order repaired and checked by reading, not compiled, not run on hardware
//
#include <hip/hip_runtime.h>

#define B_  1024
#define C_  2048
#define H_  32
#define NH_ 64
#define TMW 32

typedef __attribute__((ext_vector_type(16))) __bf16       v16bf;
typedef __attribute__((ext_vector_type(8)))  float        v8f;
typedef __attribute__((ext_vector_type(4)))  float        f32x4;
typedef __attribute__((ext_vector_type(4)))  unsigned int u32x4;
typedef __attribute__((ext_vector_type(2)))  unsigned int u32x2;

union BF16Frag { u32x4 u[2]; v16bf v; };

__device__ __forceinline__ unsigned short f2bu(float f) {
    // fp32 -> bf16 bits, round-to-nearest-even
    unsigned u = __float_as_uint(f);
    u += 0x7FFFu + ((u >> 16) & 1u);
    return (unsigned short)(u >> 16);
}

// ---------------------------------------------------------------------------
// K0: xx = shifted - x ; mx = x + xx*time_maa_x ; new_shifted = x
// ---------------------------------------------------------------------------
__global__ void prep_kernel(const float* __restrict__ x, const float* __restrict__ shifted,
                            const float* __restrict__ tmx,
                            float* __restrict__ xx, float* __restrict__ mx,
                            float* __restrict__ new_shifted) {
    int i = blockIdx.x * 256 + threadIdx.x;
    int c = i & (C_ - 1);
    float xv = x[i];
    float d  = shifted[i] - xv;
    xx[i] = d;
    mx[i] = xv + d * tmx[c];
    new_shifted[i] = xv;
}

// ---------------------------------------------------------------------------
// K1: xxx = tanh(mx @ time_maa_w1)   (1024 x 2048) @ (2048 x 160)
// ---------------------------------------------------------------------------
__global__ void maa1_kernel(const float* __restrict__ mx, const float* __restrict__ w1,
                            float* __restrict__ xxx) {
    int b = blockIdx.x;
    int j = threadIdx.x;                    // 0..159
    const float* row = mx + (size_t)b * C_;
    float acc = 0.f;
    for (int c = 0; c < C_; ++c)
        acc = fmaf(row[c], w1[c * 160 + j], acc);
    xxx[b * 160 + j] = tanhf(acc);
}

// ---------------------------------------------------------------------------
// K2: x5[f] = x + xx*(wkvrg[f] + xxx[:,f,:] @ w2[f])   (K = 32)
// ---------------------------------------------------------------------------
__global__ void maa2_kernel(const float* __restrict__ x, const float* __restrict__ xx,
                            const float* __restrict__ xxx, const float* __restrict__ w2,
                            const float* __restrict__ wkvrg, float* __restrict__ x5) {
    int c = blockIdx.x * 256 + threadIdx.x;
    int b = blockIdx.y;
    int f = blockIdx.z;
    const float* xr  = xxx + b * 160 + f * TMW;
    const float* w2f = w2 + (size_t)f * TMW * C_;
    float acc = 0.f;
#pragma unroll
    for (int e = 0; e < TMW; ++e)
        acc = fmaf(xr[e], w2f[e * C_ + c], acc);
    size_t i = (size_t)b * C_ + c;
    x5[(size_t)f * B_ * C_ + i] = x[i] + xx[i] * (wkvrg[f * C_ + c] + acc);
}

// ---------------------------------------------------------------------------
// K3: td1 = tanh(x5[0] @ time_decay_w1)    (N = 32)
// ---------------------------------------------------------------------------
__global__ void decay1_kernel(const float* __restrict__ x5, const float* __restrict__ dw1,
                              float* __restrict__ td1) {
    int b = blockIdx.x;
    int t = threadIdx.x;                    // 0..31
    const float* row = x5 + (size_t)b * C_;
    float acc = 0.f;
    for (int c = 0; c < C_; ++c)
        acc = fmaf(row[c], dw1[c * TMW + t], acc);
    td1[b * TMW + t] = tanhf(acc);
}

// ---------------------------------------------------------------------------
// K4: w = exp(-exp(td1 @ time_decay_w2 + time_decay_p))   (K = 32)
// ---------------------------------------------------------------------------
__global__ void decay2_kernel(const float* __restrict__ td1, const float* __restrict__ dw2,
                              const float* __restrict__ dp, float* __restrict__ wd) {
    int c = blockIdx.x * 256 + threadIdx.x;
    int b = blockIdx.y;
    const float* tr = td1 + b * TMW;
    float acc = dp[c];
#pragma unroll
    for (int t = 0; t < TMW; ++t)
        acc = fmaf(tr[t], dw2[t * C_ + c], acc);
    wd[(size_t)b * C_ + c] = __expf(-__expf(acc));
}

// ---------------------------------------------------------------------------
// K5: per-head RWKV state update + GroupNorm + SiLU gate.
// One wave32 per head; lane owns columns m=lane and m=lane+32.
// out_m = (sum_n r_n*tf_n*k_n)*v_m + sum_n r_n*s_nm ; ns_nm = k_n*v_m + w_n*s_nm
// ---------------------------------------------------------------------------
__global__ __launch_bounds__(256) void state_kernel(
    const float* __restrict__ rkvg, const float* __restrict__ wd,
    const float* __restrict__ state, const float* __restrict__ tf,
    const float* __restrict__ gns, const float* __restrict__ gnb,
    float* __restrict__ new_state, float* __restrict__ pre) {
    const size_t BC = (size_t)B_ * C_;
    int lane = threadIdx.x & 31;
    int wv   = threadIdx.x >> 5;
    int head = blockIdx.x * 8 + wv;         // b*H + h
    int b = head >> 5;
    int h = head & 31;
    size_t rowoff = (size_t)b * C_ + h * NH_;
    const float* r  = rkvg + 0 * BC + rowoff;
    const float* k  = rkvg + 1 * BC + rowoff;
    const float* v  = rkvg + 2 * BC + rowoff;
    const float* g  = rkvg + 3 * BC + rowoff;
    const float* wr = wd + rowoff;
    const float* tfh = tf + h * NH_;
    const float* s  = state + (size_t)head * NH_ * NH_;
    float* ns = new_state + (size_t)head * NH_ * NH_;

    __shared__ float sh[8][4][NH_];
    sh[wv][0][lane] = r[lane];   sh[wv][0][lane + 32] = r[lane + 32];
    sh[wv][1][lane] = k[lane];   sh[wv][1][lane + 32] = k[lane + 32];
    sh[wv][2][lane] = wr[lane];  sh[wv][2][lane + 32] = wr[lane + 32];
    sh[wv][3][lane] = tfh[lane]; sh[wv][3][lane + 32] = tfh[lane + 32];
    __syncthreads();

    float v0 = v[lane], v1 = v[lane + 32];
    float acc0 = 0.f, acc1 = 0.f, coef = 0.f;
#pragma unroll 4
    for (int n = 0; n < NH_; ++n) {
        float rn = sh[wv][0][n], kn = sh[wv][1][n];
        float wn = sh[wv][2][n], tn = sh[wv][3][n];
        coef = fmaf(rn * tn, kn, coef);
        float s0 = s[n * NH_ + lane];
        float s1 = s[n * NH_ + lane + 32];
        acc0 = fmaf(rn, s0, acc0);
        acc1 = fmaf(rn, s1, acc1);
        ns[n * NH_ + lane]      = fmaf(kn, v0, wn * s0);
        ns[n * NH_ + lane + 32] = fmaf(kn, v1, wn * s1);
    }
    float o0 = fmaf(coef, v0, acc0);
    float o1 = fmaf(coef, v1, acc1);

    // GroupNorm over 64 values (2 per lane) via wave32 butterflies
    float sum = o0 + o1;
    float sq  = o0 * o0 + o1 * o1;
#pragma unroll
    for (int off = 16; off >= 1; off >>= 1) {
        sum += __shfl_xor(sum, off, 32);
        sq  += __shfl_xor(sq, off, 32);
    }
    float mu  = sum * (1.f / 64.f);
    float var = sq * (1.f / 64.f) - mu * mu;
    float rs  = rsqrtf(var + 6.4e-4f);      // GN_EPS = 1e-5 * 64
    int c0 = h * NH_ + lane, c1 = c0 + 32;
    float g0 = g[lane], g1 = g[lane + 32];
    float y0 = fmaf((o0 - mu) * rs, gns[c0], gnb[c0]);
    float y1 = fmaf((o1 - mu) * rs, gns[c1], gnb[c1]);
    y0 *= g0 / (1.f + __expf(-g0));         // g * sigmoid(g)
    y1 *= g1 / (1.f + __expf(-g1));
    pre[(size_t)b * C_ + c0] = y0;
    pre[(size_t)b * C_ + c1] = y1;
}

// ---------------------------------------------------------------------------
// K6: bf16 WMMA GEMM  Out[M,N] = A[M,K] @ W[K,N], fp32 in/out, bf16 compute.
// Block 256 thr = 8 waves, block tile 128x64, K-step 32.
// Double-buffered LDS (2 x 15 KB): next tile's global_load_b128s are issued
// before the barrier and the current tile's 4 back-to-back WMMAs, then
// converted + ds_stored into the other buffer. One barrier per K-step.
// A staged as [m][k], B transposed as [n][k], row stride 40 halves (80 B =
// 20 banks) -> per-lane ds_load_b128 fragment fetches are conflict-free.
// All 4 B fragments are preloaded so the 4 WMMAs issue under one dscnt wait.
// blockIdx.z selects one of several stacked GEMMs (rkvg: 4).
// ---------------------------------------------------------------------------
__global__ __launch_bounds__(256) void wmma_gemm_kernel(
    const float* __restrict__ A, const float* __restrict__ W, float* __restrict__ Out,
    size_t aStride, size_t wStride, size_t oStride) {
    const int K = C_, NN = C_;
    A   += (size_t)blockIdx.z * aStride;
    W   += (size_t)blockIdx.z * wStride;
    Out += (size_t)blockIdx.z * oStride;
    const int n0 = blockIdx.x * 64;
    const int m0 = blockIdx.y * 128;

    __shared__ __align__(16) unsigned short Ash[2][128 * 40]; // 2 x 10.0 KB
    __shared__ __align__(16) unsigned short Bsh[2][64 * 40];  // 2 x  5.0 KB

    const int tid = threadIdx.x, lane = tid & 31, wv = tid >> 5;
    const int lm = lane & 15;
    const int a_kb = (lane >> 4) * 8;    // A frag K-half base: 0 / 8
    const int b_kb = (lane >> 4) * 8;    // B frag K-half base: 0 / 8
    const int arow = tid >> 3, ac4 = tid & 7;   // A staging: +32 rows per rep
    const int bkr  = tid >> 4, bn4 = tid & 15;  // B staging: +16 k-rows per rep

    v8f acc[4];
#pragma unroll
    for (int t = 0; t < 4; ++t)
#pragma unroll
        for (int j = 0; j < 8; ++j) acc[t][j] = 0.f;

    f32x4 ar[4], br[2];

    // ---- prologue: fetch + stage tile 0 into buffer 0 ----
#pragma unroll
    for (int i = 0; i < 4; ++i)
        ar[i] = *(const f32x4*)(A + (size_t)(m0 + arow + i * 32) * K + ac4 * 4);
#pragma unroll
    for (int i = 0; i < 2; ++i)
        br[i] = *(const f32x4*)(W + (size_t)(bkr + i * 16) * NN + n0 + bn4 * 4);
#pragma unroll
    for (int i = 0; i < 4; ++i) {
        u32x2 q;
        q.x = (unsigned)f2bu(ar[i].x) | ((unsigned)f2bu(ar[i].y) << 16);
        q.y = (unsigned)f2bu(ar[i].z) | ((unsigned)f2bu(ar[i].w) << 16);
        *(u32x2*)&Ash[0][(arow + i * 32) * 40 + ac4 * 4] = q;
    }
#pragma unroll
    for (int i = 0; i < 2; ++i) {
        int kr = bkr + i * 16;
        Bsh[0][(bn4 * 4 + 0) * 40 + kr] = f2bu(br[i].x);
        Bsh[0][(bn4 * 4 + 1) * 40 + kr] = f2bu(br[i].y);
        Bsh[0][(bn4 * 4 + 2) * 40 + kr] = f2bu(br[i].z);
        Bsh[0][(bn4 * 4 + 3) * 40 + kr] = f2bu(br[i].w);
    }

    const int NK = K / 32;
    int p = 0;
    for (int kt = 0; kt < NK; ++kt) {
        const bool more = (kt + 1) < NK;
        // ---- prefetch next tile's global data (in flight across the WMMAs) ----
        if (more) {
            const int k0 = (kt + 1) * 32;
#pragma unroll
            for (int i = 0; i < 4; ++i)
                ar[i] = *(const f32x4*)(A + (size_t)(m0 + arow + i * 32) * K + k0 + ac4 * 4);
#pragma unroll
            for (int i = 0; i < 2; ++i)
                br[i] = *(const f32x4*)(W + (size_t)(k0 + bkr + i * 16) * NN + n0 + bn4 * 4);
        }
        __syncthreads();   // buffer p fully staged (prologue / previous iteration)

        // ---- load all fragments, then 4 back-to-back WMMAs ----
        BF16Frag af, bf[4];
        const unsigned short* ap = &Ash[p][(wv * 16 + lm) * 40 + a_kb];
        af.u[0] = *(const u32x4*)(ap);
        af.u[1] = *(const u32x4*)(ap + 16);
#pragma unroll
        for (int t = 0; t < 4; ++t) {
            const unsigned short* bp = &Bsh[p][(t * 16 + lm) * 40 + b_kb];
            bf[t].u[0] = *(const u32x4*)(bp);
            bf[t].u[1] = *(const u32x4*)(bp + 16);
        }
#pragma unroll
        for (int t = 0; t < 4; ++t)
            acc[t] = __builtin_amdgcn_wmma_f32_16x16x32_bf16(
                false, af.v, false, bf[t].v, (short)0, acc[t], false, false);

        // ---- convert + stage the prefetched tile into the other buffer ----
        if (more) {
            const int q2 = p ^ 1;
#pragma unroll
            for (int i = 0; i < 4; ++i) {
                u32x2 q;
                q.x = (unsigned)f2bu(ar[i].x) | ((unsigned)f2bu(ar[i].y) << 16);
                q.y = (unsigned)f2bu(ar[i].z) | ((unsigned)f2bu(ar[i].w) << 16);
                *(u32x2*)&Ash[q2][(arow + i * 32) * 40 + ac4 * 4] = q;
            }
#pragma unroll
            for (int i = 0; i < 2; ++i) {
                int kr = bkr + i * 16;
                Bsh[q2][(bn4 * 4 + 0) * 40 + kr] = f2bu(br[i].x);
                Bsh[q2][(bn4 * 4 + 1) * 40 + kr] = f2bu(br[i].y);
                Bsh[q2][(bn4 * 4 + 2) * 40 + kr] = f2bu(br[i].z);
                Bsh[q2][(bn4 * 4 + 3) * 40 + kr] = f2bu(br[i].w);
            }
        }
        p ^= 1;
    }

    // epilogue: C/D layout — lanes<16: N=lane, M=r ; lanes>=16: N=lane-16, M=8+r
    int mrow = m0 + wv * 16 + ((lane >> 4) * 8);
#pragma unroll
    for (int t = 0; t < 4; ++t) {
        int col = n0 + t * 16 + lm;
#pragma unroll
        for (int r = 0; r < 8; ++r)
            Out[(size_t)(mrow + r) * NN + col] = acc[t][r];
    }
}

// ---------------------------------------------------------------------------
extern "C" void kernel_launch(void* const* d_in, const int* in_sizes, int n_in,
                              void* d_out, int out_size, void* d_ws, size_t ws_size,
                              hipStream_t stream) {
    const float* x      = (const float*)d_in[0];
    const float* shifted= (const float*)d_in[1];
    const float* state  = (const float*)d_in[2];
    const float* tmx    = (const float*)d_in[3];
    const float* w1     = (const float*)d_in[4];
    const float* w2     = (const float*)d_in[5];
    const float* wkvrg  = (const float*)d_in[6];
    const float* w_rkvg = (const float*)d_in[7];
    const float* dw1    = (const float*)d_in[8];
    const float* dw2    = (const float*)d_in[9];
    const float* dp     = (const float*)d_in[10];
    const float* tf     = (const float*)d_in[11];
    const float* gns    = (const float*)d_in[12];
    const float* gnb    = (const float*)d_in[13];
    const float* w_out  = (const float*)d_in[14];

    const size_t BC = (size_t)B_ * C_;
    float* out         = (float*)d_out;
    float* new_shifted = out + BC;
    float* new_state   = new_shifted + BC;

    float* ws   = (float*)d_ws;
    float* xx   = ws;                         // BC
    float* mx   = xx + BC;                    // BC
    float* xxx  = mx + BC;                    // B*160
    float* td1  = xxx + (size_t)B_ * 160;     // B*32
    float* x5   = td1 + (size_t)B_ * 32;      // 5*BC
    float* rkvg = x5 + 5 * BC;                // 4*BC
    float* wd   = rkvg + 4 * BC;              // BC
    float* pre  = wd + BC;                    // BC

    prep_kernel<<<dim3((unsigned)(BC / 256)), 256, 0, stream>>>(x, shifted, tmx, xx, mx, new_shifted);
    maa1_kernel<<<dim3(B_), 160, 0, stream>>>(mx, w1, xxx);
    maa2_kernel<<<dim3(C_ / 256, B_, 5), 256, 0, stream>>>(x, xx, xxx, w2, wkvrg, x5);
    decay1_kernel<<<dim3(B_), 32, 0, stream>>>(x5, dw1, td1);
    decay2_kernel<<<dim3(C_ / 256, B_), 256, 0, stream>>>(td1, dw2, dp, wd);
    // r,k,v,g = x5[1..4] @ w_rkvg[0..3]
    wmma_gemm_kernel<<<dim3(C_ / 64, B_ / 128, 4), 256, 0, stream>>>(
        x5 + BC, w_rkvg, rkvg, BC, (size_t)C_ * C_, BC);
    state_kernel<<<dim3(B_ * H_ / 8), 256, 0, stream>>>(
        rkvg, wd, state, tf, gns, gnb, new_state, pre);
    // out = pre @ w_out
    wmma_gemm_kernel<<<dim3(C_ / 64, B_ / 128, 1), 256, 0, stream>>>(
        pre, w_out, out, 0, 0, 0);
}
